// SpatialCorrelationSampler_5523327943026
// MI455X (gfx1250) — matrix operand order repaired and checked by reading, not compiled, not run on hardware
//
#include <hip/hip_runtime.h>

typedef __attribute__((ext_vector_type(16))) _Float16 v16h;
typedef __attribute__((ext_vector_type(8)))  _Float16 v8h;
typedef __attribute__((ext_vector_type(8)))  float    v8f;
typedef __attribute__((ext_vector_type(4)))  float    v4f;
typedef __attribute__((ext_vector_type(2)))  __fp16   v2hf;

#define B_  4
#define C_  256
#define H_  128
#define W_  128
#define PATCH 9
#define PAD 4
#define TILE 16
// Row stride in halves: 560B = 140 dwords == 12 (mod 64) -> the 16 per-half-wave
// b128 fragment reads tile all 64 LDS banks exactly; 16B aligned for ds_load_b128.
#define LSTR 280

union PK { v2hf q[2]; unsigned long long u; };

// One wave32 per (b, h, 16-wide w tile).  For each dy build the 16x32 score
// matrix S[i,j] = sum_c f1[c,w0+i]*f2[c,h+dy-4,w0-4+j] with two chains of
// v_wmma_f32_16x16x32_f16 over K=256, then scatter the 9-wide diagonal band
// S[i,i+dx] into the 81 output planes.  B tiles are double-buffered in LDS and
// staged for dy+1 underneath dy's WMMAs (clamped, branch-free hot path; zero
// padding repaired by rare scalar-branched fixups); B fragments are ds-loaded
// one kc ahead of their WMMA; dy+2's row is prefetched into WGP$/L2.
__global__ __launch_bounds__(32)
void corr81_wmma_kernel(const float* __restrict__ in1,
                        const float* __restrict__ in2,
                        float* __restrict__ out)
{
    __shared__ __align__(16) _Float16 Alds[TILE * LSTR];     // [pixel i][channel c]
    __shared__ __align__(16) _Float16 Blds[2][32 * LSTR];    // [col j][channel c], 2 bufs

    const int lane = threadIdx.x;          // 0..31
    const int w0   = blockIdx.x * TILE;
    const int h    = blockIdx.y;
    const int b    = blockIdx.z;

    const long planeHW = (long)H_ * W_;
    const float* f1  = in1 + (long)b * C_ * planeHW + (long)h * W_ + w0;
    const float* f2b = in2 + (long)b * C_ * planeHW;

    // ---- stage A tile: f1[c][w0..w0+15] -> Alds[i][c] f16, b64 stores ----
    {
        const int pixq = lane & 3;         // pixel group 0..3 (4 pixels each)
        const int chq  = lane >> 2;        // channel quad 0..7
        #pragma unroll
        for (int it = 0; it < 8; ++it) {
            const int c0 = it * 32 + chq * 4;
            const float* src = f1 + (long)c0 * planeHW + pixq * 4;
            v4f v0 = *(const v4f*)(src);
            v4f v1 = *(const v4f*)(src + planeHW);
            v4f v2 = *(const v4f*)(src + 2 * planeHW);
            v4f v3 = *(const v4f*)(src + 3 * planeHW);
            #pragma unroll
            for (int p = 0; p < 4; ++p) {
                PK pk;
                pk.q[0] = __builtin_amdgcn_cvt_pkrtz(v0[p], v1[p]);
                pk.q[1] = __builtin_amdgcn_cvt_pkrtz(v2[p], v3[p]);
                *reinterpret_cast<unsigned long long*>(
                    &Alds[(pixq * 4 + p) * LSTR + c0]) = pk.u;
            }
        }
    }

    // ---- per-lane B staging geometry (constant over dy and s) ----
    const int  colq = lane & 7;                    // column quad 0..7
    const int  chq4 = (lane >> 3) * 4;             // channel quad base 0,4,8,12
    const int  col  = w0 - PAD + colq * 4;         // may be OOB at w edges
    const int  colc = (col < 0) ? 0 : ((col > W_ - 4) ? (W_ - 4) : col); // clamped
    const float* f2col = f2b + (long)chq4 * planeHW + colc;

    // Branch-free hot path: clamped loads, unconditional packed b64 stores.
    auto stageB = [&](int nb, int s, int row) {
        const int c0 = s * 16 + chq4;
        const float* src = f2col + (long)(s * 16) * planeHW + (long)row * W_;
        v4f v0 = *(const v4f*)(src);
        v4f v1 = *(const v4f*)(src + planeHW);
        v4f v2 = *(const v4f*)(src + 2 * planeHW);
        v4f v3 = *(const v4f*)(src + 3 * planeHW);
        #pragma unroll
        for (int p = 0; p < 4; ++p) {
            PK pk;
            pk.q[0] = __builtin_amdgcn_cvt_pkrtz(v0[p], v1[p]);
            pk.q[1] = __builtin_amdgcn_cvt_pkrtz(v2[p], v3[p]);
            *reinterpret_cast<unsigned long long*>(
                &Blds[nb][(colq * 4 + p) * LSTR + c0]) = pk.u;
        }
    };

    // Rare fixup: zero rows [j0, j0+nrows) of a B buffer (channels 0..255).
    auto zeroRows = [&](int nb, int j0, int nrows) {
        for (int t = lane; t < nrows * 64; t += 32) {
            const int j  = j0 + (t >> 6);
            const int c0 = (t & 63) * 4;
            *reinterpret_cast<unsigned long long*>(
                &Blds[nb][j * LSTR + c0]) = 0ULL;
        }
    };
    // Only cols 0..23 feed the diagonal band; cols 24..31 may hold garbage.
    // Left edge tile: cols j=0..3 are pad; right edge tile: j=20..23 are pad.
    const bool edgeL = (w0 == 0);
    const bool edgeR = (w0 + TILE + PAD > W_);
    auto fixB = [&](int nb, bool rowok) {
        if (!rowok)      zeroRows(nb, 0, 24);
        else if (edgeL)  zeroRows(nb, 0, 4);
        else if (edgeR)  zeroRows(nb, 20, 4);
    };

    // ---- preload all 8 A fragments (K=0..255), reused for all 9 dy ----
    // A operand k order used here (same as B): lane (m = lane&15, hf = lane>>4) holds
    // 16 consecutive K at kc*32 + hf*16
    const int m  = lane & 15;
    const int hf = lane >> 4;
    v16h afrag[8];
    #pragma unroll
    for (int kc = 0; kc < 8; ++kc) {
        const _Float16* ap = &Alds[m * LSTR + kc * 32 + hf * 16];
        v8h lo = *(const v8h*)(ap);
        v8h hi = *(const v8h*)(ap + 8);
        afrag[kc] = __builtin_shufflevector(lo, hi, 0,1,2,3,4,5,6,7,
                                                    8,9,10,11,12,13,14,15);
    }

    // ---- prologue: stage B(dy=0) into buffer 0 ----
    {
        const int  row0  = h - PAD;                // always <= 127
        const bool r0ok  = (row0 >= 0);
        if (r0ok) {
            #pragma unroll
            for (int s = 0; s < 16; ++s) stageB(0, s, row0);
        }
        fixB(0, r0ok);
    }

    // B fragment loader: chain0 (cols 0..15) and chain1 (cols 16..31).
    // B operand k order used here (same as A): lane (n = lane&15, kh = lane>>4) holds
    // 16 consecutive K at kc*32 + kh*16.
    const int nrow_base = (lane & 15) * LSTR + hf * 16;
    auto loadB = [&](int cb, int kc, v8h* u0, v8h* u1) {
        const _Float16* bp0 = &Blds[cb][nrow_base + kc * 32];
        u0[0] = *(const v8h*)(bp0);
        u0[1] = *(const v8h*)(bp0 + 8);
        const _Float16* bp1 = bp0 + 16 * LSTR;
        u1[0] = *(const v8h*)(bp1);
        u1[1] = *(const v8h*)(bp1 + 8);
    };

    float* outb = out + (long)b * (PATCH * PATCH) * planeHW + (long)h * W_ + w0;
    const int ncol = lane & 15;

    for (int dy = 0; dy < PATCH; ++dy) {
        const int  cb      = dy & 1, nb = cb ^ 1;
        const int  nrow    = h + (dy + 1) - PAD;
        const bool dostage = (dy + 1 < PATCH);
        const bool nrowok  = (nrow >= 0) && (nrow < H_);
        const int  prow    = h + (dy + 2) - PAD;
        const bool pok     = (dy + 2 < PATCH) && (prow >= 0) && (prow < H_);
        const int  pcol    = (w0 >= PAD) ? (w0 - PAD) : 0;

        v8f acc0 = {}; v8f acc1 = {};
        v8h c0f[2], c1f[2], n0f[2], n1f[2];
        loadB(cb, 0, c0f, c1f);
        #pragma unroll
        for (int kc = 0; kc < 8; ++kc) {
            if (kc < 7) loadB(cb, kc + 1, n0f, n1f);      // one kc ahead
            if (dostage && nrowok) {                       // scalar branch
                stageB(nb, 2 * kc,     nrow);
                stageB(nb, 2 * kc + 1, nrow);
            }
            if (pok) {                                     // warm WGP$/L2, dy+2
                const float* pf = f2b + (long)(kc * 32 + lane) * planeHW
                                      + (long)prow * W_ + pcol;
                __builtin_prefetch(pf, 0, 1);
            }
            v16h b0 = __builtin_shufflevector(c0f[0], c0f[1], 0,1,2,3,4,5,6,7,
                                              8,9,10,11,12,13,14,15);
            v16h b1 = __builtin_shufflevector(c1f[0], c1f[1], 0,1,2,3,4,5,6,7,
                                              8,9,10,11,12,13,14,15);
            acc0 = __builtin_amdgcn_wmma_f32_16x16x32_f16(
                       false, afrag[kc], false, b0, (short)0, acc0, false, false);
            acc1 = __builtin_amdgcn_wmma_f32_16x16x32_f16(
                       false, afrag[kc], false, b1, (short)0, acc1, false, false);
            c0f[0] = n0f[0]; c0f[1] = n0f[1];
            c1f[0] = n1f[0]; c1f[1] = n1f[1];
        }
        if (dostage) fixB(nb, nrowok);                     // rare scalar fixups

        // ---- scatter diagonal band: corr[i,dx] = S[i, i+dx], dx in [0,8] ----
        // C/D layout: VGPR r, lane L -> M = r + 8*(L>>4), N = L&15 (+16 chain 1).
        // dx0 and dx1 windows are mutually exclusive -> one store per r.
        #pragma unroll
        for (int r = 0; r < 8; ++r) {
            const int  i   = r + 8 * hf;
            const int  dx0 = ncol - i;
            const int  dx1 = dx0 + 16;
            const bool u0  = (unsigned)dx0 <= 8u;
            const bool u1  = (unsigned)dx1 <= 8u;
            const int  dx  = u0 ? dx0 : dx1;
            const float v  = u0 ? acc0[r] : acc1[r];
            if (u0 || u1)
                outb[(long)(dy * PATCH + dx) * planeHW + i] = v;
        }
    }
}

extern "C" void kernel_launch(void* const* d_in, const int* in_sizes, int n_in,
                              void* d_out, int out_size, void* d_ws, size_t ws_size,
                              hipStream_t stream) {
    const float* in1 = (const float*)d_in[0];
    const float* in2 = (const float*)d_in[1];
    float* outp = (float*)d_out;
    (void)in_sizes; (void)n_in; (void)out_size; (void)d_ws; (void)ws_size;

    dim3 grid(W_ / TILE, H_, B_);   // 8 x 128 x 4 = 4096 waves
    corr81_wmma_kernel<<<grid, 32, 0, stream>>>(in1, in2, outp);
}
